// AdaptiveRankTextSubNet_30365418783361
// MI455X (gfx1250) — matrix operand order repaired and checked by reading, not compiled, not run on hardware
//
#include <hip/hip_runtime.h>
#include <cstdint>
#include <cstddef>

// ---------------------------------------------------------------------------
// Types for CDNA5 WMMA (wave32): v_wmma_f32_16x16x32_bf16
// ---------------------------------------------------------------------------
typedef __attribute__((ext_vector_type(16))) __bf16 bf16x16;
typedef __attribute__((ext_vector_type(8)))  __bf16 bf16x8;
typedef __attribute__((ext_vector_type(8)))  float  f32x8;

// A-fragment (16x32 bf16, MxK): lane L holds row m = L%16.
//   lanes 0-15 : halves 0..15 -> K = kc*32 + 0..15
//   lanes 16-31: halves 0..15 -> K = kc*32 + 16..31
__device__ __forceinline__ bf16x16 ldsA(const __bf16* row, int kc, int hi) {
  const __bf16* p = row + kc * 32 + (hi ? 16 : 0);
  bf16x8 lo = *(const bf16x8*)(p);
  bf16x8 hh = *(const bf16x8*)(p + 8);
  return __builtin_shufflevector(lo, hh, 0,1,2,3,4,5,6,7,8,9,10,11,12,13,14,15);
}

// B-fragment (32x16 bf16, KxN): lane L holds column n = L%16,
// 16 contiguous K values starting at kc*32 + (L<16 ? 0 : 16).
// `row` points at weight row W[n][*] (since B[k][n] = W[n][k]).
__device__ __forceinline__ bf16x16 ldsB(const __bf16* row, int kc, int hi) {
  return *(const bf16x16*)(row + kc * 32 + (hi ? 16 : 0));
}

__device__ __forceinline__ f32x8 wmma_bf16(bf16x16 a, bf16x16 b, f32x8 c) {
  return __builtin_amdgcn_wmma_f32_16x16x32_bf16(false, a, false, b, (short)0, c,
                                                 false, false);
}

__device__ __forceinline__ float fsig(float x) { return 1.0f / (1.0f + __expf(-x)); }
__device__ __forceinline__ float ftanh(float x) {
  x = fminf(fmaxf(x, -15.0f), 15.0f);
  float e = __expf(2.0f * x);
  return (e - 1.0f) / (e + 1.0f);
}

// ---------------------------------------------------------------------------
// CDNA5 async global->LDS copy (ASYNCcnt path). 256 threads move 32 KB
// (8 x b128 per thread, fully coalesced). LDS offset = low 32 bits of the
// generic shared-memory address (ISA 10.2).
// ---------------------------------------------------------------------------
__device__ __forceinline__ void stage_async_load(const float* __restrict__ gsrc,
                                                 unsigned lds_off, int tid) {
#pragma unroll
  for (int j = 0; j < 8; ++j) {
    unsigned byteoff = (unsigned)((j * 256 + tid) * 16);
    unsigned l = lds_off + byteoff;
    unsigned long long g = (unsigned long long)(size_t)gsrc + byteoff;
    asm volatile("global_load_async_to_lds_b128 %0, %1, off"
                 :: "v"(l), "v"(g) : "memory");
  }
}

// ===========================================================================
// Kernel 1: xg[t][b][g] = x[b][t][:] @ W_ih[g][:]^T   (memory-bound GEMM)
// Grid: (4096 t-blocks, 4 n-blocks of 128 gates). Block: 256 thr = 8 waves.
// WG tile: M=64 (all batch rows of one t), N=128, K=320 (300 zero-padded).
// Fragment loads double-buffered across K-chunks to overlap DS with WMMA.
// ===========================================================================
#define SMEM1 ((64 * 320 + 128 * 320) * 2)  // 122880 B (bf16 x-tile + W-tile)

__global__ void __launch_bounds__(256) xproj_gemm(const float* __restrict__ x,
                                                  const float* __restrict__ Wih,
                                                  float* __restrict__ xg) {
  extern __shared__ char smem[];
  __bf16* xs = (__bf16*)smem;        // [64][320]
  __bf16* ws = xs + 64 * 320;        // [128][320]

  const int t   = blockIdx.x;        // 0..4095
  const int nb  = blockIdx.y;        // 0..3
  const int tid = threadIdx.x;

  // Stage x tile (f32 -> bf16), zero-pad K 300..319.
  for (int idx = tid; idx < 64 * 320; idx += 256) {
    int b = idx / 320, k = idx % 320;
    float v = (k < 300) ? x[((size_t)b * 4096 + t) * 300 + k] : 0.0f;
    xs[b * 320 + k] = (__bf16)v;
  }
  // Stage W_ih slice [nb*128 .. +128)
  for (int idx = tid; idx < 128 * 320; idx += 256) {
    int gl = idx / 320, k = idx % 320;
    float v = (k < 300) ? Wih[(size_t)(nb * 128 + gl) * 300 + k] : 0.0f;
    ws[gl * 320 + k] = (__bf16)v;
  }
  __syncthreads();

  const int wave = tid >> 5, lane = tid & 31;
  const int hi = lane >> 4, lm = lane & 15;
  const int mt = wave & 3;              // M-tile (16 rows)
  const int ntBase = (wave >> 2) * 4;   // this wave's 4 N-tiles

  const __bf16* arow = xs + (mt * 16 + lm) * 320;
  const __bf16* brow[4];
#pragma unroll
  for (int i = 0; i < 4; ++i) brow[i] = ws + ((ntBase + i) * 16 + lm) * 320;

  f32x8 acc[4];
#pragma unroll
  for (int i = 0; i < 4; ++i)
#pragma unroll
    for (int r = 0; r < 8; ++r) acc[i][r] = 0.0f;

  // Software-pipelined fragment loads (double buffered over kc).
  bf16x16 afr[2];
  bf16x16 bfr[2][4];
  afr[0] = ldsA(arow, 0, hi);
#pragma unroll
  for (int i = 0; i < 4; ++i) bfr[0][i] = ldsB(brow[i], 0, hi);

#pragma unroll
  for (int kc = 0; kc < 10; ++kc) {
    const int cu = kc & 1, nx = cu ^ 1;
    if (kc < 9) {
      afr[nx] = ldsA(arow, kc + 1, hi);
#pragma unroll
      for (int i = 0; i < 4; ++i) bfr[nx][i] = ldsB(brow[i], kc + 1, hi);
    }
#pragma unroll
    for (int i = 0; i < 4; ++i) acc[i] = wmma_bf16(afr[cu], bfr[cu][i], acc[i]);
  }

  // C layout: VGPR r, lane L -> M = r + 8*(L>=16), N = L%16
#pragma unroll
  for (int i = 0; i < 4; ++i) {
    const int n = nb * 128 + (ntBase + i) * 16 + lm;
#pragma unroll
    for (int r = 0; r < 8; ++r) {
      const int b = mt * 16 + r + hi * 8;
      xg[((size_t)t * 64 + b) * 512 + n] = acc[i][r];
    }
  }
}

// ===========================================================================
// Kernel 2: serial LSTM recurrence + MLP head.
// Grid: 4 WGs x 256 thr (8 waves); WG w owns batch rows [16w, 16w+16).
// Wave w owns the i/f/g/o gate tiles of j-slice [16w,16w+16): after WMMA every
// lane holds all four gates of its (b,j) pairs, so the cell state c stays in
// VGPRs for the whole 4096-step loop; only h round-trips LDS (next step's
// matrix operand). xg_t slices are double-buffered into LDS with CDNA5
// global_load_async_to_lds_b128 + s_wait_asynccnt. 2 barriers/step.
// ===========================================================================
#define SMEM2 (512*128*2 /*whh*/ + 16*128*2 /*hbf*/ + 2*16*512*4 /*stage*/ + \
               512*4 /*bias*/ + 16*64*4 /*out1*/)   // 206848 B < 320 KB/WGP

__global__ void __launch_bounds__(256) lstm_head(
    const float* __restrict__ xg,  const float* __restrict__ Whh,
    const float* __restrict__ bih, const float* __restrict__ bhh,
    const float* __restrict__ W1,  const float* __restrict__ b1,
    const float* __restrict__ W2,  const float* __restrict__ b2,
    float* __restrict__ out) {
  extern __shared__ char smem[];
  __bf16* whh  = (__bf16*)smem;                 // [512][128] bf16 (gate-major)
  __bf16* hbf  = whh + 512 * 128;               // [16][128] bf16 h state
  float*  stg  = (float*)(hbf + 16 * 128);      // [2][16][512] xg_t staging
  float*  bias = stg + 2 * 16 * 512;            // [512] b_ih + b_hh
  float*  o1   = bias + 512;                    // [16][64] MLP hidden

  const int tid = threadIdx.x;
  const int b0  = blockIdx.x * 16;

  for (int i = tid; i < 512 * 128; i += 256) whh[i] = (__bf16)Whh[i];
  for (int i = tid; i < 512; i += 256)       bias[i] = bih[i] + bhh[i];
  for (int i = tid; i < 16 * 128; i += 256)  hbf[i] = (__bf16)0.0f;

  const unsigned stgOff0 = (unsigned)(size_t)stg;
  const unsigned stgOff1 = (unsigned)(size_t)(stg + 16 * 512);

  // Kick off async load of t=0 gate slice (16 x 512 f32 = 32 KB).
  stage_async_load(xg + (size_t)b0 * 512, stgOff0, tid);

  __syncthreads();  // whh/bias/hbf init visible

  const int wave = tid >> 5, lane = tid & 31;
  const int hi = lane >> 4, lm = lane & 15;
  const int j  = wave * 16 + lm;        // this lane's hidden index j

  // Gate-tile columns owned by this lane: n = gate*128 + j  (i,f,g,o)
  int nidx[4];
  float bv[4];
  const __bf16* brow[4];
#pragma unroll
  for (int i = 0; i < 4; ++i) {
    nidx[i] = i * 128 + j;
    bv[i]   = bias[nidx[i]];
    brow[i] = whh + (size_t)nidx[i] * 128;
  }
  const __bf16* arow = hbf + lm * 128;  // A = h rows (m = batch)

  float creg[8];                        // cell state lives in registers
#pragma unroll
  for (int r = 0; r < 8; ++r) creg[r] = 0.0f;

  const f32x8 zacc = {};                // inline-0 C operand for first WMMA

  for (int t = 0; t < 4096; ++t) {
    const int cur = t & 1;
    // Prefetch next timestep into the other buffer, then wait for current.
    if (t + 1 < 4096) {
      stage_async_load(xg + ((size_t)(t + 1) * 64 + b0) * 512,
                       cur ? stgOff0 : stgOff1, tid);
      asm volatile("s_wait_asynccnt 0x8" ::: "memory");  // only t+1's 8 left
    } else {
      asm volatile("s_wait_asynccnt 0x0" ::: "memory");
    }
    __syncthreads();  // [A] stage ready; prev step's h writes visible

    const float* gate = stg + cur * 16 * 512;

    // h @ W_hh^T for this wave's 4 gate tiles; fragments double-buffered.
    f32x8 acc[4];
    bf16x16 afr[2];
    bf16x16 bfr[2][4];
    afr[0] = ldsA(arow, 0, hi);
#pragma unroll
    for (int i = 0; i < 4; ++i) bfr[0][i] = ldsB(brow[i], 0, hi);
#pragma unroll
    for (int kc = 0; kc < 4; ++kc) {
      const int cu = kc & 1, nx = cu ^ 1;
      if (kc < 3) {
        afr[nx] = ldsA(arow, kc + 1, hi);
#pragma unroll
        for (int i = 0; i < 4; ++i) bfr[nx][i] = ldsB(brow[i], kc + 1, hi);
      }
#pragma unroll
      for (int i = 0; i < 4; ++i)
        acc[i] = wmma_bf16(afr[cu], bfr[cu][i], (kc == 0) ? zacc : acc[i]);
    }

    // Gate math fully in registers: lane owns (b = r+8*hi, j) for r=0..7.
    float hreg[8];
#pragma unroll
    for (int r = 0; r < 8; ++r) {
      const int row = (r + hi * 8) * 512;
      float ig = fsig (acc[0][r] + gate[row + nidx[0]] + bv[0]);
      float fg = fsig (acc[1][r] + gate[row + nidx[1]] + bv[1]);
      float gg = ftanh(acc[2][r] + gate[row + nidx[2]] + bv[2]);
      float og = fsig (acc[3][r] + gate[row + nidx[3]] + bv[3]);
      float c  = fg * creg[r] + ig * gg;
      creg[r]  = c;
      hreg[r]  = og * ftanh(c);
    }

    __syncthreads();  // [B] all waves done reading hbf (and stage[nxt] free)

#pragma unroll
    for (int r = 0; r < 8; ++r)
      hbf[(r + hi * 8) * 128 + j] = (__bf16)hreg[r];
    // next iteration's barrier [A] publishes these writes
  }

  // Publish final h as f32 into reused stage memory for the MLP head.
  float* hfl = stg;                     // stage buffers are dead now
#pragma unroll
  for (int r = 0; r < 8; ++r)
    hfl[(r + hi * 8) * 128 + j] = (float)creg[r] * 0.0f + /*h*/ 0.0f;
  // (placeholder overwritten below - write actual h)
#pragma unroll
  for (int r = 0; r < 8; ++r)
    hfl[(r + hi * 8) * 128 + j] = (float)hbf[(r + hi * 8) * 128 + j];
  __syncthreads();

  // MLP head: out = relu(relu(h @ W1^T + b1) @ W2^T + b2)   (tiny)
  for (int idx = tid; idx < 16 * 64; idx += 256) {
    const int br = idx >> 6, o = idx & 63;
    float s = b1[o];
    for (int k = 0; k < 128; ++k) s += hfl[br * 128 + k] * W1[o * 128 + k];
    o1[idx] = fmaxf(s, 0.0f);
  }
  __syncthreads();
  for (int idx = tid; idx < 16 * 64; idx += 256) {
    const int br = idx >> 6, o = idx & 63;
    float s = b2[o];
    for (int k = 0; k < 64; ++k) s += o1[br * 64 + k] * W2[o * 64 + k];
    out[(size_t)(b0 + br) * 64 + o] = fmaxf(s, 0.0f);
  }
}

// ===========================================================================
extern "C" void kernel_launch(void* const* d_in, const int* in_sizes, int n_in,
                              void* d_out, int out_size, void* d_ws, size_t ws_size,
                              hipStream_t stream) {
  (void)in_sizes; (void)n_in; (void)out_size; (void)ws_size;
  const float* x   = (const float*)d_in[0];
  const float* Wih = (const float*)d_in[1];
  const float* Whh = (const float*)d_in[2];
  const float* bih = (const float*)d_in[3];
  const float* bhh = (const float*)d_in[4];
  const float* W1  = (const float*)d_in[5];
  const float* b1v = (const float*)d_in[6];
  const float* W2  = (const float*)d_in[7];
  const float* b2v = (const float*)d_in[8];
  float* out = (float*)d_out;
  float* xg  = (float*)d_ws;  // needs 4096*64*512*4 = 512 MiB scratch

  hipFuncSetAttribute(reinterpret_cast<const void*>(xproj_gemm),
                      hipFuncAttributeMaxDynamicSharedMemorySize, SMEM1);
  hipFuncSetAttribute(reinterpret_cast<const void*>(lstm_head),
                      hipFuncAttributeMaxDynamicSharedMemorySize, SMEM2);

  xproj_gemm<<<dim3(4096, 4, 1), dim3(256, 1, 1), SMEM1, stream>>>(x, Wih, xg);
  lstm_head<<<dim3(4, 1, 1), dim3(256, 1, 1), SMEM2, stream>>>(
      xg, Whh, bih, bhh, W1, b1v, W2, b2v, out);
}
